// SelfAttention2D_43911745634356
// MI455X (gfx1250) — matrix operand order repaired and checked by reading, not compiled, not run on hardware
//
#include <hip/hip_runtime.h>
#include <hip/hip_bf16.h>

// ---------------------------------------------------------------------------
// SelfAttention2D for MI455X (gfx1250), wave32 + v_wmma_f32_16x16x32_bf16.
// All matmuls run on the WMMA pipe with fp32 accumulation; BN+ReLU fused into
// GEMM epilogues; attention kernel stages its tiles into LDS via the Tensor
// Data Mover (tensor_load_to_lds + s_wait_tensorcnt) when available.
// ---------------------------------------------------------------------------

#define DEV __device__ __forceinline__

typedef __attribute__((ext_vector_type(16))) __bf16 v16bf;
typedef __attribute__((ext_vector_type(8)))  float  v8f;
typedef __attribute__((ext_vector_type(4)))  unsigned int v4u;
typedef __attribute__((ext_vector_type(8)))  int    v8i;
typedef __attribute__((ext_vector_type(4)))  int    v4i;

union BF8 { uint4 u; __bf16 h[8]; };

#define REGUL_F 0.08838834764831845f   // sqrt(2/256)
#define BN_EPS_F 1e-3f

// ---- WMMA fragment helpers (layouts per CDNA5 ISA §7.12.2) -----
// A (16x32 bf16, M x K): lane&15 = row M, lane>>4 picks K-groups {kg*8..+7}
// and {16+kg*8..+7}; two contiguous 16B runs per lane.
DEV v16bf loadA_bf16(const __bf16* A, int lda, int row0, int k0, int lane) {
  const int r = lane & 15, kg = lane >> 4;
  const __bf16* p = A + (size_t)(row0 + r) * lda + k0 + kg * 8;
  BF8 lo, hi;
  lo.u = *(const uint4*)(p);
  hi.u = *(const uint4*)(p + 16);
  v16bf a;
#pragma unroll
  for (int i = 0; i < 8; ++i) { a[i] = lo.h[i]; a[i + 8] = hi.h[i]; }
  return a;
}

// B (32x16 bf16, K x N): lane&15 = column N, lane>>4 picks K-groups
// {kg*8..+7} and {16+kg*8..+7}; strided scalar loads down the column.
DEV v16bf loadB_bf16(const __bf16* Bm, int ldb, int k0, int n0, int lane) {
  const int n = lane & 15, kg = lane >> 4;
  const __bf16* p = Bm + (size_t)(k0 + kg * 8) * ldb + n0 + n;
  v16bf b;
#pragma unroll
  for (int i = 0; i < 8; ++i) {
    b[i] = p[(size_t)i * ldb];
    b[i + 8] = p[(size_t)(i + 16) * ldb];
  }
  return b;
}

DEV v8f wmma_bf16(v16bf a, v16bf b, v8f c) {
  // (neg_a, A, neg_b, B, c_mod, C, reuse_a, reuse_b)
  return __builtin_amdgcn_wmma_f32_16x16x32_bf16(false, a, false, b,
                                                 (short)0, c, false, false);
}

// ---- Tensor Data Mover: 2D tile (rows x cols bf16, row-major, stride=cols)
// D# per CDNA5 ISA. Issued once per calling wave.
// This toolchain exposes the 6-arg builtin: (v4u g0, v8i g1, v4i, v4i, v8i, cpol)
DEV void tdm_load_2d_bf16(const __bf16* gsrc, __bf16* ldst, int rows, int cols) {
#if __has_builtin(__builtin_amdgcn_tensor_load_to_lds)
  const unsigned long long ga = (unsigned long long)(uintptr_t)(const void*)gsrc;
  const unsigned int la = (unsigned int)(uintptr_t)(void*)ldst;  // addr[31:0] = LDS offset
  v4u g0;
  g0[0] = 1u;                                            // count=1, user desc
  g0[1] = la;                                            // lds_addr [63:32]
  g0[2] = (unsigned int)ga;                              // global_addr [95:64]
  g0[3] = ((unsigned int)(ga >> 32) & 0x01FFFFFFu) |     // global_addr [120:96]
          (2u << 30);                                    // type=2 ("image")
  v8i g1;
  g1[0] = 0x00010000;                 // wg_mask=0, data_size=1 (2 bytes)
  g1[1] = (cols & 0xFFFF) << 16;      // tensor_dim0[15:0]  @ bits 63:48
  g1[2] = ((cols >> 16) & 0xFFFF) |   // tensor_dim0[31:16] @ bits 79:64
          ((rows & 0xFFFF) << 16);    // tensor_dim1[15:0]  @ bits 95:80
  g1[3] = ((rows >> 16) & 0xFFFF) |   // tensor_dim1[31:16] @ bits 111:96
          ((cols & 0xFFFF) << 16);    // tile_dim0          @ bits 127:112
  g1[4] = rows & 0xFFFF;              // tile_dim1 @ bits 143:128, tile_dim2=0
  g1[5] = cols;                       // tensor_dim0_stride[31:0]
  g1[6] = 0;
  g1[7] = 0;
  const v4i gz4 = {0, 0, 0, 0};       // 2D tensor: groups 2/3 disabled
  const v8i gz8 = {0, 0, 0, 0, 0, 0, 0, 0};
  __builtin_amdgcn_tensor_load_to_lds(g0, g1, gz4, gz4, gz8, 0);
#else
  // fallback: plain copy by this wave (correct, slower)
  const int lane = threadIdx.x & 31;
  const uint4* s = (const uint4*)gsrc;
  uint4* d = (uint4*)ldst;
  for (int i = lane; i < rows * cols / 8; i += 32) d[i] = s[i];
#endif
}

DEV void tdm_wait_all() {
#if __has_builtin(__builtin_amdgcn_tensor_load_to_lds)
  __builtin_amdgcn_s_wait_tensorcnt(0);
#endif
}

// ---- tiny prep kernels ----------------------------------------------------
__global__ void sa2d_cvt_bf16_kernel(const float* __restrict__ src,
                                     __bf16* __restrict__ dst, int n) {
  int i = blockIdx.x * blockDim.x + threadIdx.x;
  if (i < n) dst[i] = (__bf16)src[i];
}

__global__ void sa2d_bnprep_kernel(const float* __restrict__ g,
                                   const float* __restrict__ b,
                                   const float* __restrict__ m,
                                   const float* __restrict__ v,
                                   float2* __restrict__ out, int n) {
  int i = blockIdx.x * blockDim.x + threadIdx.x;
  if (i < n) {
    float s = g[i] * rsqrtf(v[i] + BN_EPS_F);
    out[i] = make_float2(s, b[i] - m[i] * s);
  }
}

// ---- GEMM 1: X[N x 256] (f32) @ W[256 x 128] (bf16) -> bf16, BN+ReLU opt --
// block = 256 threads (8 waves) covers 16 rows x 128 cols. The 16x256 fp32
// A-tile is converted to bf16 ONCE into LDS (instead of 8x per-wave), then
// each wave owns one 16x16 output tile; K unrolled 256/32 = 8 WMMAs.
__global__ __launch_bounds__(256) void sa2d_gemm_in_kernel(
    const float* __restrict__ X, const __bf16* __restrict__ Wt,
    __bf16* __restrict__ Y, const float2* __restrict__ bn, int relu) {
  __shared__ __bf16 sX[16 * 256];  // 8KB
  const int tid = threadIdx.x;
  const int lane = tid & 31, wave = tid >> 5;
  const int row0 = blockIdx.x * 16, n0 = wave * 16;

  // cooperative f32->bf16 staging: 4096 elems, 8 per thread per pass
  {
    const float* src = X + (size_t)row0 * 256;
#pragma unroll
    for (int i = tid * 8; i < 16 * 256; i += 256 * 8) {
      float4 a0 = *(const float4*)(src + i);
      float4 a1 = *(const float4*)(src + i + 4);
      BF8 d;
      d.h[0] = (__bf16)a0.x; d.h[1] = (__bf16)a0.y;
      d.h[2] = (__bf16)a0.z; d.h[3] = (__bf16)a0.w;
      d.h[4] = (__bf16)a1.x; d.h[5] = (__bf16)a1.y;
      d.h[6] = (__bf16)a1.z; d.h[7] = (__bf16)a1.w;
      *(uint4*)(sX + i) = d.u;
    }
  }
  __syncthreads();

  v8f acc = {};
#pragma unroll
  for (int k0 = 0; k0 < 256; k0 += 32) {
    v16bf a = loadA_bf16(sX, 256, 0, k0, lane);
    v16bf b = loadB_bf16(Wt, 128, k0, n0, lane);
    acc = wmma_bf16(a, b, acc);
  }
  // D layout: vgpr j, lane -> (m = j + (lane>>4)*8, n = lane&15)
  const int mh = lane >> 4, n = lane & 15, c = n0 + n;
  float s = 1.f, t = 0.f;
  if (bn) { float2 st = bn[c]; s = st.x; t = st.y; }
#pragma unroll
  for (int j = 0; j < 8; ++j) {
    float val = acc[j] * s + t;
    if (relu) val = fmaxf(val, 0.f);
    Y[(size_t)(row0 + j + mh * 8) * 128 + c] = (__bf16)val;
  }
}

// ---- GEMM 2: H[N x 128] (bf16) @ W[128 x 128] -> bf16, BN+ReLU ------------
__global__ __launch_bounds__(256) void sa2d_gemm_h_kernel(
    const __bf16* __restrict__ X, const __bf16* __restrict__ Wt,
    __bf16* __restrict__ Y, const float2* __restrict__ bn) {
  const int lane = threadIdx.x & 31, wave = threadIdx.x >> 5;
  const int row0 = blockIdx.x * 16, n0 = wave * 16;
  v8f acc = {};
#pragma unroll
  for (int k0 = 0; k0 < 128; k0 += 32) {
    __builtin_prefetch(X + (size_t)(row0 + (lane & 15)) * 128 + k0 + 32, 0, 0);
    v16bf a = loadA_bf16(X, 128, row0, k0, lane);
    v16bf b = loadB_bf16(Wt, 128, k0, n0, lane);
    acc = wmma_bf16(a, b, acc);
  }
  const int mh = lane >> 4, n = lane & 15, c = n0 + n;
  const float2 st = bn[c];
#pragma unroll
  for (int j = 0; j < 8; ++j) {
    float val = fmaxf(acc[j] * st.x + st.y, 0.f);
    Y[(size_t)(row0 + j + mh * 8) * 128 + c] = (__bf16)val;
  }
}

// ---- fused row attention: one workgroup per (b,h) -------------------------
// theta/gamma rows (32KB each) staged into LDS by the Tensor Data Mover while
// all 8 waves build phi-transposed in LDS; then S = theta@phiT via WMMA,
// register softmax (shfl_xor butterflies over 16-lane halves, matching the
// C/D fragment layout), P back through LDS, O = P@gamma via WMMA.
__global__ __launch_bounds__(256) void sa2d_attn_kernel(
    const __bf16* __restrict__ Theta, const __bf16* __restrict__ Phi,
    const __bf16* __restrict__ Gamma, __bf16* __restrict__ Out) {
  extern __shared__ char smem_raw[];
  __bf16* sP    = (__bf16*)smem_raw;        // theta, later reused as P
  __bf16* sPhiT = sP    + 128 * 128;        // [c][v]
  __bf16* sGam  = sPhiT + 128 * 128;        // [v][c]
  const int tid = threadIdx.x;
  const int lane = tid & 31, wave = tid >> 5;
  const size_t base = (size_t)blockIdx.x * 128 * 128;

  // TDM: one descriptor per 128x128 bf16 tile, issued by wave 0 only
  if (wave == 0) {
    tdm_load_2d_bf16(Theta + base, sP, 128, 128);
    tdm_load_2d_bf16(Gamma + base, sGam, 128, 128);
  }
  // meanwhile: all threads build phi^T (scatter stores; uint4 = 8 bf16)
  {
    const uint4* gP = (const uint4*)(Phi + base);
    for (int i = tid; i < 128 * 128 / 8; i += 256) {
      BF8 d; d.u = gP[i];
      const int v = i >> 4, c0 = (i & 15) * 8;
#pragma unroll
      for (int j = 0; j < 8; ++j) sPhiT[(c0 + j) * 128 + v] = d.h[j];
    }
  }
  if (wave == 0) tdm_wait_all();  // TENSORcnt==0 -> theta/gamma in LDS
  __syncthreads();

  // S = theta @ phiT  (16 queries x 128 keys per wave)
  const int row0 = wave * 16;
  v8f S[8];
#pragma unroll
  for (int t = 0; t < 8; ++t) S[t] = (v8f){};
#pragma unroll
  for (int k0 = 0; k0 < 128; k0 += 32) {
    v16bf a = loadA_bf16(sP, 128, row0, k0, lane);
#pragma unroll
    for (int t = 0; t < 8; ++t) {
      v16bf b = loadB_bf16(sPhiT, 128, k0, t * 16, lane);
      S[t] = wmma_bf16(a, b, S[t]);
    }
  }

  // softmax over 128 keys per query row: reduce 8 tiles per-lane, then
  // xor-butterfly (masks<16 keep the two 16-lane halves independent).
#pragma unroll
  for (int j = 0; j < 8; ++j) {
    float mx = -1e30f;
#pragma unroll
    for (int t = 0; t < 8; ++t) mx = fmaxf(mx, S[t][j]);
    for (int x = 8; x >= 1; x >>= 1) mx = fmaxf(mx, __shfl_xor(mx, x, 32));
    float sum = 0.f;
#pragma unroll
    for (int t = 0; t < 8; ++t) {
      float e = __expf((S[t][j] - mx) * REGUL_F);
      S[t][j] = e;
      sum += e;
    }
    for (int x = 8; x >= 1; x >>= 1) sum += __shfl_xor(sum, x, 32);
    const float inv = 1.f / sum;
#pragma unroll
    for (int t = 0; t < 8; ++t) S[t][j] *= inv;
  }

  // write P (bf16) into this wave's own 16 theta rows (rows are private per
  // wave: each wave reads only its own 16 rows of sP -> no barrier needed)
  {
    const int mh = lane >> 4, n = lane & 15;
#pragma unroll
    for (int t = 0; t < 8; ++t)
#pragma unroll
      for (int j = 0; j < 8; ++j)
        sP[(size_t)(row0 + j + mh * 8) * 128 + t * 16 + n] = (__bf16)S[t][j];
  }

  // O = P @ gamma
  v8f O[8];
#pragma unroll
  for (int t = 0; t < 8; ++t) O[t] = (v8f){};
#pragma unroll
  for (int k0 = 0; k0 < 128; k0 += 32) {
    v16bf a = loadA_bf16(sP, 128, row0, k0, lane);
#pragma unroll
    for (int t = 0; t < 8; ++t) {
      v16bf b = loadB_bf16(sGam, 128, k0, t * 16, lane);
      O[t] = wmma_bf16(a, b, O[t]);
    }
  }
  {
    const int mh = lane >> 4, n = lane & 15;
#pragma unroll
    for (int t = 0; t < 8; ++t)
#pragma unroll
      for (int j = 0; j < 8; ++j)
        Out[base + (size_t)(row0 + j + mh * 8) * 128 + t * 16 + n] =
            (__bf16)O[t][j];
  }
}

// ---- final GEMM: attn[N x 128] (bf16) @ rw[128 x 256] -> f32 d_out --------
__global__ __launch_bounds__(256) void sa2d_gemm_out_kernel(
    const __bf16* __restrict__ X, const __bf16* __restrict__ Wt,
    float* __restrict__ Y) {
  const int lane = threadIdx.x & 31, wave = threadIdx.x >> 5;
  const int row0 = blockIdx.x * 16;
  const int n0 = (blockIdx.y * 8 + wave) * 16;
  v8f acc = {};
#pragma unroll
  for (int k0 = 0; k0 < 128; k0 += 32) {
    v16bf a = loadA_bf16(X, 128, row0, k0, lane);
    v16bf b = loadB_bf16(Wt, 256, k0, n0, lane);
    acc = wmma_bf16(a, b, acc);
  }
  const int mh = lane >> 4, n = lane & 15;
#pragma unroll
  for (int j = 0; j < 8; ++j)
    Y[(size_t)(row0 + j + mh * 8) * 256 + n0 + n] = acc[j];
}

// ---------------------------------------------------------------------------
extern "C" void kernel_launch(void* const* d_in, const int* in_sizes, int n_in,
                              void* d_out, int out_size, void* d_ws,
                              size_t ws_size, hipStream_t stream) {
  (void)in_sizes; (void)n_in; (void)out_size; (void)ws_size;
  constexpr int Bn = 8, Hh = 128, Ww = 128, Ff = 256, FH = 128;
  constexpr int N = Bn * Hh * Ww;  // 131072 pixels

  const float* inputs = (const float*)d_in[0];
  const float* tw1 = (const float*)d_in[1];
  const float* tw2 = (const float*)d_in[2];
  const float* pw1 = (const float*)d_in[3];
  const float* pw2 = (const float*)d_in[4];
  const float* gw  = (const float*)d_in[5];
  const float* rw  = (const float*)d_in[6];
  const float* t1p[4] = {(const float*)d_in[7],  (const float*)d_in[8],
                         (const float*)d_in[9],  (const float*)d_in[10]};
  const float* t2p[4] = {(const float*)d_in[11], (const float*)d_in[12],
                         (const float*)d_in[13], (const float*)d_in[14]};
  const float* p1p[4] = {(const float*)d_in[15], (const float*)d_in[16],
                         (const float*)d_in[17], (const float*)d_in[18]};
  const float* p2p[4] = {(const float*)d_in[19], (const float*)d_in[20],
                         (const float*)d_in[21], (const float*)d_in[22]};

  char* ws = (char*)d_ws;
  __bf16* tw1b = (__bf16*)(ws + 0);         // 256*128
  __bf16* tw2b = (__bf16*)(ws + 65536);     // 128*128
  __bf16* pw1b = (__bf16*)(ws + 98304);     // 256*128
  __bf16* pw2b = (__bf16*)(ws + 163840);    // 128*128
  __bf16* gwb  = (__bf16*)(ws + 196608);    // 256*128
  __bf16* rwb  = (__bf16*)(ws + 262144);    // 128*256
  float2* bnT1 = (float2*)(ws + 327680);
  float2* bnT2 = (float2*)(ws + 328704);
  float2* bnP1 = (float2*)(ws + 329728);
  float2* bnP2 = (float2*)(ws + 330752);
  const size_t MB32 = (size_t)N * FH * 2;
  __bf16* hth   = (__bf16*)(ws + (1u << 20) + 0 * MB32);
  __bf16* hph   = (__bf16*)(ws + (1u << 20) + 1 * MB32);
  __bf16* theta = (__bf16*)(ws + (1u << 20) + 2 * MB32);
  __bf16* phi   = (__bf16*)(ws + (1u << 20) + 3 * MB32);
  __bf16* gamma = (__bf16*)(ws + (1u << 20) + 4 * MB32);
  __bf16* attn  = (__bf16*)(ws + (1u << 20) + 5 * MB32);

  // --- prep: convert weights, fold BN ---
  sa2d_cvt_bf16_kernel<<<128, 256, 0, stream>>>(tw1, tw1b, Ff * FH);
  sa2d_cvt_bf16_kernel<<<64, 256, 0, stream>>>(tw2, tw2b, FH * FH);
  sa2d_cvt_bf16_kernel<<<128, 256, 0, stream>>>(pw1, pw1b, Ff * FH);
  sa2d_cvt_bf16_kernel<<<64, 256, 0, stream>>>(pw2, pw2b, FH * FH);
  sa2d_cvt_bf16_kernel<<<128, 256, 0, stream>>>(gw, gwb, Ff * FH);
  sa2d_cvt_bf16_kernel<<<128, 256, 0, stream>>>(rw, rwb, FH * Ff);
  sa2d_bnprep_kernel<<<1, 128, 0, stream>>>(t1p[0], t1p[1], t1p[2], t1p[3], bnT1, FH);
  sa2d_bnprep_kernel<<<1, 128, 0, stream>>>(t2p[0], t2p[1], t2p[2], t2p[3], bnT2, FH);
  sa2d_bnprep_kernel<<<1, 128, 0, stream>>>(p1p[0], p1p[1], p1p[2], p1p[3], bnP1, FH);
  sa2d_bnprep_kernel<<<1, 128, 0, stream>>>(p2p[0], p2p[1], p2p[2], p2p[3], bnP2, FH);

  const dim3 blk(256);
  // --- projections ---
  sa2d_gemm_in_kernel<<<dim3(N / 16), blk, 0, stream>>>(inputs, tw1b, hth, bnT1, 1);
  sa2d_gemm_in_kernel<<<dim3(N / 16), blk, 0, stream>>>(inputs, pw1b, hph, bnP1, 1);
  sa2d_gemm_in_kernel<<<dim3(N / 16), blk, 0, stream>>>(inputs, gwb, gamma, nullptr, 0);
  sa2d_gemm_h_kernel<<<dim3(N / 16), blk, 0, stream>>>(hth, tw2b, theta, bnT2);
  sa2d_gemm_h_kernel<<<dim3(N / 16), blk, 0, stream>>>(hph, pw2b, phi, bnP2);
  // --- fused row attention (softmax over W=128), 96KB dynamic LDS ---
  sa2d_attn_kernel<<<dim3(Bn * Hh), blk, 3 * 128 * 128 * 2, stream>>>(theta, phi,
                                                                      gamma, attn);
  // --- output projection ---
  sa2d_gemm_out_kernel<<<dim3(N / 16, 2), blk, 0, stream>>>(attn, rwb,
                                                            (float*)d_out);
}
